// _BiMamba3_13529146982967
// MI455X (gfx1250) — matrix operand order repaired and checked by reading, not compiled, not run on hardware
//
#include <hip/hip_runtime.h>
#include <stdint.h>
#include <stddef.h>

// ---------------- problem dimensions ----------------
#define LSEQ    4096
#define BATCH   16
#define MTOK    (BATCH * LSEQ)   // 65536 tokens
#define DMODEL  128
#define DSTATE  64
#define HEADDIM 64
#define DINNER  256
#define NHEADS  4
#define CONVDIM 384
#define NPROJ   644              // real in_proj cols
#define NPAD    768              // padded to 48*16 (and 6*128)
#define EPS     1e-5f

typedef __bf16 bf16_t;
typedef __attribute__((ext_vector_type(16))) __bf16    v16bf;
typedef __attribute__((ext_vector_type(8)))  float     v8f;
typedef __attribute__((ext_vector_type(4)))  uint32_t  v4u;
typedef __attribute__((ext_vector_type(8)))  int       v8i;
typedef __attribute__((ext_vector_type(4)))  int       v4i;
typedef __attribute__((ext_vector_type(4)))  uint32_t  u32x4;

// ---------------- scalar bf16 helpers (native cvt on gfx1250) ---------------
__device__ __forceinline__ bf16_t f2bf(float f)  { return (bf16_t)f; }
__device__ __forceinline__ float  bf2f(bf16_t b) { return (float)b; }

// ---------------- WMMA fragment helpers (wave32, 16x16x32 bf16) -------------
// A matrix 16x32 bf16: lane L -> row = L&15, half h = L>>4,
//   elements [0..7] = K 8h..8h+7, [8..15] = K 16+8h..16+8h+7.
__device__ __forceinline__ v16bf load_frag_a(const bf16_t* src, int ld) {
    const int lane = threadIdx.x & 31;
    const int row  = lane & 15;
    const int h    = lane >> 4;
    const bf16_t* p = src + (size_t)row * ld + 8 * h;
    union { v16bf v; u32x4 q[2]; } u;
    u.q[0] = *(const u32x4*)(p);
    u.q[1] = *(const u32x4*)(p + 16);
    return u.v;
}
// B matrix 32x16 bf16 stored column-major as [n][K] rows (i.e. B^T row-major):
// lane L -> col n = L&15, elements [0..7] = K 8h..8h+7, [8..15] = K 16+8h..16+8h+7.
__device__ __forceinline__ v16bf load_frag_b(const bf16_t* src, int ld) {
    const int lane = threadIdx.x & 31;
    const int col  = lane & 15;
    const int h    = lane >> 4;
    const bf16_t* p = src + (size_t)col * ld + 8 * h;
    union { v16bf v; u32x4 q[2]; } u;
    u.q[0] = *(const u32x4*)(p);
    u.q[1] = *(const u32x4*)(p + 16);
    return u.v;
}
__device__ __forceinline__ v8f wmma_bf16(v16bf a, v16bf b, v8f c) {
    return __builtin_amdgcn_wmma_f32_16x16x32_bf16(
        /*neg_a=*/false, a, /*neg_b=*/false, b,
        /*c_mod=*/(short)0, c, /*reuse_a=*/false, /*reuse_b=*/false);
}

// ---------------- Tensor Data Mover: 2-D bf16 tile -> LDS -------------------
// D# built per CDNA5 ISA (group0: count/lds/global/type,
// group1: data_size=2B, tensor dims, tile dims, dim0 stride). Groups 2/3 zero
// (<=2-D tensor). 6-arg builtin:
//   (uint32x4 g0, int32x8 g1, int32x4 g2, int32x4 g3, int32x8 g4, i32 cpol)
// Issued by one wave, completion via TENSORcnt.
__device__ __forceinline__ void tdm_load_2d_to_lds(uint32_t lds_byte_off,
                                                   const void* gptr,
                                                   uint32_t rows,
                                                   uint32_t row_elems,
                                                   uint32_t row_stride_elems) {
    uint64_t ga = (uint64_t)(uintptr_t)gptr;
    v4u g0;
    g0.x = 1u;                                            // count=1, user mode
    g0.y = lds_byte_off;                                  // lds_addr
    g0.z = (uint32_t)(ga & 0xFFFFFFFFu);                  // global_addr lo
    g0.w = (uint32_t)((ga >> 32) & 0x01FFFFFFu) | (2u << 30); // addr hi | type=2
    v8i g1;
    g1[0] = (int)(1u << 16);                              // data_size=1 -> 2 bytes
    g1[1] = (int)((row_elems & 0xFFFFu) << 16);           // tensor_dim0 lo16
    g1[2] = (int)(((row_elems >> 16) & 0xFFFFu) |
                  ((rows & 0xFFFFu) << 16));              // dim0 hi | dim1 lo
    g1[3] = (int)(((rows >> 16) & 0xFFFFu) |
                  ((row_elems & 0xFFFFu) << 16));         // dim1 hi | tile_dim0
    g1[4] = (int)(rows & 0xFFFFu);                        // tile_dim1 (tile_dim2=0)
    g1[5] = (int)row_stride_elems;                        // tensor_dim0_stride lo32
    g1[6] = 0;
    g1[7] = 0;
    v4i z4 = {0, 0, 0, 0};
    v8i z8 = {0, 0, 0, 0, 0, 0, 0, 0};
    __builtin_amdgcn_tensor_load_to_lds(g0, g1, z4, z4, z8, 0);
}

// ============================================================================
// K0a: x fp32 -> bf16 (optionally sequence-reversed per batch)
// ============================================================================
__global__ void convert_x_kernel(const float* __restrict__ x,
                                 bf16_t* __restrict__ xbf, int rev) {
    int idx = blockIdx.x * blockDim.x + threadIdx.x;           // MTOK*DMODEL
    int r = idx >> 7, ch = idx & 127;
    int b = r >> 12, l = r & 4095;
    int sr = rev ? ((b << 12) + (LSEQ - 1 - l)) : r;
    xbf[idx] = f2bf(x[(size_t)sr * DMODEL + ch]);
}
// K0b: in_proj (128 x 644) -> transposed, zero-padded bf16 (768 x 128)
__global__ void convert_win_kernel(const float* __restrict__ W,
                                   bf16_t* __restrict__ Wt) {
    int idx = blockIdx.x * blockDim.x + threadIdx.x;           // NPAD*DMODEL
    int n = idx >> 7, k = idx & 127;
    Wt[idx] = (n < NPROJ) ? f2bf(W[(size_t)k * NPROJ + n]) : f2bf(0.f);
}
// K0c: out_proj (256 x 128) -> transposed bf16 (128 x 256)
__global__ void convert_wout_kernel(const float* __restrict__ W,
                                    bf16_t* __restrict__ Wt) {
    int idx = blockIdx.x * blockDim.x + threadIdx.x;           // DMODEL*DINNER
    int n = idx >> 8, k = idx & 255;
    Wt[idx] = f2bf(W[(size_t)k * DMODEL + n]);
}

// ============================================================================
// K1: zxbcdt = x @ in_proj   (M=65536, K=128, N=768 padded), bf16 out.
// Full weight (768x128 bf16 = 192KB) staged in LDS via TDM once per block.
// ============================================================================
__global__ __launch_bounds__(256)
void gemm_in_wmma(const bf16_t* __restrict__ A,    // MTOK x 128
                  const bf16_t* __restrict__ Wt,   // NPAD x 128 (n-major)
                  bf16_t* __restrict__ Z) {        // MTOK x NPAD
    extern __shared__ bf16_t sW[];                 // NPAD*DMODEL
    const int wave = threadIdx.x >> 5;
    const int lane = threadIdx.x & 31;
    if (wave == 0) {
        tdm_load_2d_to_lds(0u, Wt, NPAD, DMODEL, DMODEL);
        __builtin_amdgcn_s_wait_tensorcnt(0);
    }
    __syncthreads();
    const int m0 = blockIdx.x * 128 + wave * 16;
    v16bf afrag[4];
#pragma unroll
    for (int ks = 0; ks < 4; ++ks)
        afrag[ks] = load_frag_a(A + (size_t)m0 * DMODEL + ks * 32, DMODEL);
    for (int nt = 0; nt < NPAD / 16; ++nt) {
        v8f acc = {0.f, 0.f, 0.f, 0.f, 0.f, 0.f, 0.f, 0.f};
#pragma unroll
        for (int ks = 0; ks < 4; ++ks)
            acc = wmma_bf16(afrag[ks],
                            load_frag_b(sW + nt * 16 * DMODEL + ks * 32, DMODEL),
                            acc);
        const int col   = nt * 16 + (lane & 15);
        const int rbase = m0 + 8 * (lane >> 4);
#pragma unroll
        for (int j = 0; j < 8; ++j)
            Z[(size_t)(rbase + j) * NPAD + col] = f2bf(acc[j]);
    }
}

// ============================================================================
// K2: depthwise conv(K=4) + SiLU + split (xh | B | C), dt=softplus(+bias),
//     dA = dt * (-exp(A_log)).  One thread per (token, channel), 388 ch/token.
// ============================================================================
__global__ void conv_silu_kernel(const bf16_t* __restrict__ Z,
                                 const float* __restrict__ conv_w,  // 384 x 4
                                 const float* __restrict__ conv_b,
                                 const float* __restrict__ A_log,
                                 const float* __restrict__ dt_bias,
                                 bf16_t* __restrict__ xh,   // M x 256
                                 bf16_t* __restrict__ Bm,   // M x 64
                                 bf16_t* __restrict__ Cm,   // M x 64
                                 float* __restrict__ dtv,   // M x 4
                                 float* __restrict__ dAv) { // M x 4
    long idx = (long)blockIdx.x * blockDim.x + threadIdx.x;
    int r  = (int)(idx / 388);
    int ch = (int)(idx % 388);
    if (r >= MTOK) return;
    int l = r & 4095;
    if (ch < CONVDIM) {
        float acc = conv_b[ch];
#pragma unroll
        for (int i = 0; i < 4; ++i) {
            int ls = l - 3 + i;
            if (ls >= 0)
                acc += bf2f(Z[(size_t)(r - 3 + i) * NPAD + DINNER + ch]) *
                       conv_w[ch * 4 + i];
        }
        float s = acc / (1.f + __expf(-acc));                // SiLU
        bf16_t v = f2bf(s);
        if (ch < DINNER)            xh[(size_t)r * DINNER + ch] = v;
        else if (ch < DINNER + 64)  Bm[(size_t)r * 64 + (ch - DINNER)] = v;
        else                        Cm[(size_t)r * 64 + (ch - DINNER - 64)] = v;
    } else {
        int h = ch - CONVDIM;
        float d  = bf2f(Z[(size_t)r * NPAD + 640 + h]) + dt_bias[h];
        float sp = (d > 20.f) ? d : log1pf(__expf(d));       // softplus
        dtv[(size_t)r * 4 + h] = sp;
        dAv[(size_t)r * 4 + h] = sp * (-__expf(A_log[h]));
    }
}

// ============================================================================
// K3a: intra-chunk SSD. One block per (b, chunk, head). 8 waves, WMMA 64x64x64.
//   G = C @ B^T; T = exp(seg(dA)) .* G;  Y_diag = T @ (x*dt) + D*x
//   states = B^T @ (exp(Acs_end - Acs) .* x*dt)   (stored [p][n], bf16)
// ============================================================================
__global__ __launch_bounds__(256)
void ssd_intra_kernel(const bf16_t* __restrict__ xh,
                      const bf16_t* __restrict__ Bm,
                      const bf16_t* __restrict__ Cm,
                      const float* __restrict__ dtv,
                      const float* __restrict__ dAv,
                      const float* __restrict__ Dvec,
                      float* __restrict__ yb,       // M x 256 fp32
                      bf16_t* __restrict__ states,  // [b][h][c][p][n] bf16
                      float* __restrict__ cdec,     // [b][h][c]
                      float* __restrict__ acsg) {   // M x 4
    __shared__ float  sAcs[64];
    __shared__ bf16_t sT [64 * 64];   // [l][s]  masked decayed C.B^T
    __shared__ bf16_t sXT[64 * 64];   // [p][l]  x*dt transposed
    __shared__ bf16_t sXD[64 * 64];   // [p][l]  x*dt * decay_to_end
    __shared__ bf16_t sBT[64 * 64];   // [n][l]  B transposed
    const int tid  = threadIdx.x;
    const int wave = tid >> 5, lane = tid & 31;
    const int h = blockIdx.x & 3;
    const int c = (blockIdx.x >> 2) & 63;
    const int b = blockIdx.x >> 8;
    const int r0 = b * LSEQ + c * 64;

    if (tid == 0) {
        float acc = 0.f;
        for (int l = 0; l < 64; ++l) { acc += dAv[(size_t)(r0 + l) * 4 + h]; sAcs[l] = acc; }
    }
    __syncthreads();
    const float acsLast = sAcs[63];
    if (tid < 64) acsg[(size_t)(r0 + tid) * 4 + h] = sAcs[tid];

    for (int e = tid; e < 4096; e += 256) {
        int l = e >> 6, p = e & 63;
        float dt_l = dtv[(size_t)(r0 + l) * 4 + h];
        float xv = bf2f(xh[(size_t)(r0 + l) * DINNER + h * 64 + p]) * dt_l;
        sXT[p * 64 + l] = f2bf(xv);
        sXD[p * 64 + l] = f2bf(xv * __expf(acsLast - sAcs[l]));
        sBT[p * 64 + l] = Bm[(size_t)(r0 + l) * 64 + p]; // reuse (l,p)->(n=p)
    }
    __syncthreads();

    // ---- G = C @ B^T, mask lower-tri, apply exp(Acs[l]-Acs[s]) -> sT ----
    for (int t = wave; t < 16; t += 8) {
        int mt = t >> 2, nt = t & 3;
        v8f acc = {0.f, 0.f, 0.f, 0.f, 0.f, 0.f, 0.f, 0.f};
#pragma unroll
        for (int ks = 0; ks < 2; ++ks)
            acc = wmma_bf16(load_frag_a(Cm + (size_t)(r0 + mt * 16) * 64 + ks * 32, 64),
                            load_frag_b(Bm + (size_t)(r0 + nt * 16) * 64 + ks * 32, 64),
                            acc);
        int scol = nt * 16 + (lane & 15);
#pragma unroll
        for (int j = 0; j < 8; ++j) {
            int lrow = mt * 16 + j + 8 * (lane >> 4);
            float v = (lrow >= scol) ? acc[j] * __expf(sAcs[lrow] - sAcs[scol]) : 0.f;
            sT[lrow * 64 + scol] = f2bf(v);
        }
    }
    __syncthreads();

    // ---- Y_diag = sT @ (x*dt)  (+ D skip) -> yb ----
    const float Dh = Dvec[h];
    for (int t = wave; t < 16; t += 8) {
        int mt = t >> 2, pt = t & 3;
        v8f acc = {0.f, 0.f, 0.f, 0.f, 0.f, 0.f, 0.f, 0.f};
#pragma unroll
        for (int ks = 0; ks < 2; ++ks)
            acc = wmma_bf16(load_frag_a(sT + mt * 16 * 64 + ks * 32, 64),
                            load_frag_b(sXT + pt * 16 * 64 + ks * 32, 64), acc);
        int pcol = pt * 16 + (lane & 15);
#pragma unroll
        for (int j = 0; j < 8; ++j) {
            int lrow = mt * 16 + j + 8 * (lane >> 4);
            size_t rr = (size_t)(r0 + lrow);
            float skip = Dh * bf2f(xh[rr * DINNER + h * 64 + pcol]);
            yb[rr * DINNER + h * 64 + pcol] = acc[j] + skip;
        }
    }

    // ---- states = B^T @ (decayed x*dt) -> states[b][h][c][p][n] ----
    const size_t sbase = ((size_t)(b * NHEADS + h) * 64 + c) * 4096;
    for (int t = wave; t < 16; t += 8) {
        int ntile = t >> 2, pt = t & 3;
        v8f acc = {0.f, 0.f, 0.f, 0.f, 0.f, 0.f, 0.f, 0.f};
#pragma unroll
        for (int ks = 0; ks < 2; ++ks)
            acc = wmma_bf16(load_frag_a(sBT + ntile * 16 * 64 + ks * 32, 64),
                            load_frag_b(sXD + pt * 16 * 64 + ks * 32, 64), acc);
        int pcol = pt * 16 + (lane & 15);
#pragma unroll
        for (int j = 0; j < 8; ++j) {
            int nrow = ntile * 16 + j + 8 * (lane >> 4);
            states[sbase + (size_t)pcol * 64 + nrow] = f2bf(acc[j]);
        }
    }
    if (tid == 0) cdec[(b * NHEADS + h) * 64 + c] = __expf(acsLast);
}

// ============================================================================
// K3b: inter-chunk scan (sequential over 64 chunks), in-place states -> prev.
// ============================================================================
__global__ __launch_bounds__(256)
void ssd_scan_kernel(bf16_t* __restrict__ states, const float* __restrict__ cdec) {
    const int bh = blockIdx.x;       // b*NHEADS + h
    const int tid = threadIdx.x;
    float carry[16];
#pragma unroll
    for (int i = 0; i < 16; ++i) carry[i] = 0.f;
    for (int c = 0; c < 64; ++c) {
        const float dec = cdec[bh * 64 + c];
        const size_t base = ((size_t)bh * 64 + c) * 4096;
#pragma unroll
        for (int i = 0; i < 16; ++i) {
            int e = i * 256 + tid;
            float st = bf2f(states[base + e]);
            states[base + e] = f2bf(carry[i]);     // prev for this chunk
            carry[i] = st + carry[i] * dec;
        }
    }
}

// ============================================================================
// K3c: Y_off = exp(Acs) .* (C @ prev^T), accumulate into yb.
// ============================================================================
__global__ __launch_bounds__(256)
void ssd_off_kernel(const bf16_t* __restrict__ Cm,
                    const bf16_t* __restrict__ prev,   // [b][h][c][p][n]
                    const float* __restrict__ acsg,
                    float* __restrict__ yb) {
    const int tid  = threadIdx.x;
    const int wave = tid >> 5, lane = tid & 31;
    const int h = blockIdx.x & 3;
    const int c = (blockIdx.x >> 2) & 63;
    const int b = blockIdx.x >> 8;
    const int r0 = b * LSEQ + c * 64;
    const size_t pbase = ((size_t)(b * NHEADS + h) * 64 + c) * 4096;
    for (int t = wave; t < 16; t += 8) {
        int mt = t >> 2, pt = t & 3;
        v8f acc = {0.f, 0.f, 0.f, 0.f, 0.f, 0.f, 0.f, 0.f};
#pragma unroll
        for (int ks = 0; ks < 2; ++ks)
            acc = wmma_bf16(load_frag_a(Cm + (size_t)(r0 + mt * 16) * 64 + ks * 32, 64),
                            load_frag_b(prev + pbase + (size_t)pt * 16 * 64 + ks * 32, 64),
                            acc);
        int pcol = pt * 16 + (lane & 15);
#pragma unroll
        for (int j = 0; j < 8; ++j) {
            int lrow = mt * 16 + j + 8 * (lane >> 4);
            size_t rr = (size_t)(r0 + lrow);
            float scale = __expf(acsg[rr * 4 + h]);
            yb[rr * DINNER + h * 64 + pcol] += scale * acc[j];
        }
    }
}

// ============================================================================
// K4: gate with SiLU(z) + RMSNorm -> bf16 for out-proj GEMM.
// ============================================================================
__global__ __launch_bounds__(256)
void gate_norm_kernel(const float* __restrict__ yb,
                      const bf16_t* __restrict__ Z,      // z = cols [0,256)
                      const float* __restrict__ norm_w,
                      bf16_t* __restrict__ ynorm) {
    __shared__ float red[256];
    const int r = blockIdx.x, ch = threadIdx.x;
    float yv = yb[(size_t)r * DINNER + ch];
    float z  = bf2f(Z[(size_t)r * NPAD + ch]);
    float yg = yv * (z / (1.f + __expf(-z)));
    red[ch] = yg * yg;
    __syncthreads();
#pragma unroll
    for (int s = 128; s > 0; s >>= 1) {
        if (ch < s) red[ch] += red[ch + s];
        __syncthreads();
    }
    float inv = rsqrtf(red[0] / (float)DINNER + EPS);
    ynorm[(size_t)r * DINNER + ch] = f2bf(yg * inv * norm_w[ch]);
}

// ============================================================================
// K5: out = ynorm @ out_proj (K=256, N=128). TDM-staged weight in LDS (64KB).
// bwd direction: row-reversed accumulate into d_out (no separate combine).
// ============================================================================
__global__ __launch_bounds__(256)
void gemm_out_wmma(const bf16_t* __restrict__ A,    // MTOK x 256
                   const bf16_t* __restrict__ Wt,   // 128 x 256 (n-major)
                   float* __restrict__ Out,         // MTOK x 128
                   int accumulate, int reverse) {
    extern __shared__ bf16_t sW[];                  // DMODEL*DINNER
    const int wave = threadIdx.x >> 5;
    const int lane = threadIdx.x & 31;
    if (wave == 0) {
        tdm_load_2d_to_lds(0u, Wt, DMODEL, DINNER, DINNER);
        __builtin_amdgcn_s_wait_tensorcnt(0);
    }
    __syncthreads();
    const int m0 = blockIdx.x * 128 + wave * 16;
    v16bf afrag[8];
#pragma unroll
    for (int ks = 0; ks < 8; ++ks)
        afrag[ks] = load_frag_a(A + (size_t)m0 * DINNER + ks * 32, DINNER);
    for (int nt = 0; nt < DMODEL / 16; ++nt) {
        v8f acc = {0.f, 0.f, 0.f, 0.f, 0.f, 0.f, 0.f, 0.f};
#pragma unroll
        for (int ks = 0; ks < 8; ++ks)
            acc = wmma_bf16(afrag[ks],
                            load_frag_b(sW + nt * 16 * DINNER + ks * 32, DINNER),
                            acc);
        const int col   = nt * 16 + (lane & 15);
        const int rbase = m0 + 8 * (lane >> 4);
#pragma unroll
        for (int j = 0; j < 8; ++j) {
            int r = rbase + j;
            int bb = r >> 12, l = r & 4095;
            int orow = reverse ? ((bb << 12) + (LSEQ - 1 - l)) : r;
            size_t o = (size_t)orow * DMODEL + col;
            if (accumulate) Out[o] += acc[j];
            else            Out[o]  = acc[j];
        }
    }
}

// ============================================================================
// Host launcher
// ============================================================================
extern "C" void kernel_launch(void* const* d_in, const int* in_sizes, int n_in,
                              void* d_out, int out_size, void* d_ws, size_t ws_size,
                              hipStream_t stream) {
    (void)in_sizes; (void)n_in; (void)out_size; (void)ws_size;
    const float* x = (const float*)d_in[0];
    float* out = (float*)d_out;
    char* ws = (char*)d_ws;
    size_t off = 0;
    auto take = [&](size_t bytes) -> char* {
        char* p = ws + off;
        off += (bytes + 255) & ~(size_t)255;
        return p;
    };
    // single-direction scratch, reused for fwd then bwd (~275 MB total)
    bf16_t* xbf    = (bf16_t*)take((size_t)MTOK * DMODEL * 2);     //  16 MB
    bf16_t* WtA    = (bf16_t*)take((size_t)NPAD * DMODEL * 2);     // 192 KB
    bf16_t* WtO    = (bf16_t*)take((size_t)DMODEL * DINNER * 2);   //  64 KB
    bf16_t* Zbuf   = (bf16_t*)take((size_t)MTOK * NPAD * 2);       // 100 MB
    bf16_t* xh     = (bf16_t*)take((size_t)MTOK * DINNER * 2);     //  32 MB
    bf16_t* Bm     = (bf16_t*)take((size_t)MTOK * 64 * 2);         //   8 MB
    bf16_t* Cm     = (bf16_t*)take((size_t)MTOK * 64 * 2);         //   8 MB
    float*  dtv    = (float*)take((size_t)MTOK * 4 * 4);           //   1 MB
    float*  dAv    = (float*)take((size_t)MTOK * 4 * 4);           //   1 MB
    float*  acsg   = (float*)take((size_t)MTOK * 4 * 4);           //   1 MB
    float*  yb     = (float*)take((size_t)MTOK * DINNER * 4);      //  64 MB
    bf16_t* states = (bf16_t*)take((size_t)BATCH * NHEADS * 64 * 4096 * 2); // 32 MB
    float*  cdec   = (float*)take((size_t)BATCH * NHEADS * 64 * 4);
    bf16_t* ynorm  = states;  // reuse: states dead after ssd_off (same size)

    for (int d = 0; d < 2; ++d) {
        const float* in_proj  = (const float*)d_in[1 + 8 * d + 0];
        const float* conv_w   = (const float*)d_in[1 + 8 * d + 1];
        const float* conv_b   = (const float*)d_in[1 + 8 * d + 2];
        const float* A_log    = (const float*)d_in[1 + 8 * d + 3];
        const float* dt_bias  = (const float*)d_in[1 + 8 * d + 4];
        const float* Dvec     = (const float*)d_in[1 + 8 * d + 5];
        const float* norm_w   = (const float*)d_in[1 + 8 * d + 6];
        const float* out_proj = (const float*)d_in[1 + 8 * d + 7];

        convert_x_kernel<<<(MTOK * DMODEL) / 256, 256, 0, stream>>>(x, xbf, d);
        convert_win_kernel<<<(NPAD * DMODEL) / 256, 256, 0, stream>>>(in_proj, WtA);
        convert_wout_kernel<<<(DMODEL * DINNER) / 256, 256, 0, stream>>>(out_proj, WtO);

        gemm_in_wmma<<<MTOK / 128, 256, (size_t)NPAD * DMODEL * 2, stream>>>(xbf, WtA, Zbuf);

        conv_silu_kernel<<<((long)MTOK * 388 + 255) / 256, 256, 0, stream>>>(
            Zbuf, conv_w, conv_b, A_log, dt_bias, xh, Bm, Cm, dtv, dAv);

        ssd_intra_kernel<<<BATCH * 64 * NHEADS, 256, 0, stream>>>(
            xh, Bm, Cm, dtv, dAv, Dvec, yb, states, cdec, acsg);

        ssd_scan_kernel<<<BATCH * NHEADS, 256, 0, stream>>>(states, cdec);

        ssd_off_kernel<<<BATCH * 64 * NHEADS, 256, 0, stream>>>(Cm, states, acsg, yb);

        gate_norm_kernel<<<MTOK, 256, 0, stream>>>(yb, Zbuf, norm_w, ynorm);

        gemm_out_wmma<<<MTOK / 128, 256, (size_t)DMODEL * DINNER * 2, stream>>>(
            ynorm, WtO, out, /*accumulate=*/d, /*reverse=*/d);
    }
}
